// DecoderStepKV_14053132992916
// MI455X (gfx1250) — matrix operand order repaired and checked by reading, not compiled, not run on hardware
//
#include <hip/hip_runtime.h>
#include <hip/hip_bf16.h>
#include <math.h>

// ---------------- Types ----------------
typedef __attribute__((ext_vector_type(16))) _Float16 v16h;
typedef __attribute__((ext_vector_type(4)))  _Float16 v4h;
typedef __attribute__((ext_vector_type(8)))  float    v8f;

#define NS 384
#define NH 6
#define DH 64
#define BATCH 32
#define NLAYER 4
#define CACHE_T 447
#define FULL_T 448
#define CROSS_T 1500
#define VOCAB 51865

__device__ __forceinline__ float gelu_exact(float x) {
    return 0.5f * x * (1.0f + erff(x * 0.70710678118654752f));
}

// ---------------- Generic WMMA GEMM: out[32,N] = act(A[32,K] @ W + bias) + res
// W is [K,N] when transB==0, [N,K] when transB==1 (both fp32 row-major).
// Block: 128 threads (4 waves). Each block computes a 32 x 64 output tile.
// Wave w handles columns [w*16, w*16+16) of the tile, two 16x16 accumulators.
// LDS layouts chosen so every WMMA fragment is contiguous per lane:
//   As[m][k]  : A frag = 8+8 consecutive halfs  -> ds_load_b128 x2 per reg block
//   Bs[n][k]  : B frag = 8+8 consecutive halfs   -> ds_load_b128 x2
// Staging uses float4 (global_load_b128) everywhere; all tile rows are
// 16B-aligned because every K, and N of a [K,N] W, is a multiple of 64.
__global__ void wmma_gemm32_kernel(const float* __restrict__ A,
                                   const float* __restrict__ W,
                                   const float* __restrict__ bias,
                                   const float* __restrict__ res,
                                   float* __restrict__ out,
                                   int K, int N, int transB, int act) {
    __shared__ _Float16 As[32][72];   // pitch 144B: 16B-aligned fragments
    __shared__ _Float16 Bs[64][72];   // stored TRANSPOSED: Bs[n][k]

    const int tid  = threadIdx.x;
    const int wv   = tid >> 5;       // wave id 0..3
    const int lane = tid & 31;
    const int r    = lane & 15;      // sub-lane row/col index
    const int hi   = lane >> 4;      // half-wave
    const int nl   = wv * 16 + r;    // local output column for this lane
    const long long nbase = (long long)blockIdx.x * 64;
    const float4 zero4 = {0.f, 0.f, 0.f, 0.f};

    v8f c0 = {0.f,0.f,0.f,0.f,0.f,0.f,0.f,0.f};
    v8f c1 = {0.f,0.f,0.f,0.f,0.f,0.f,0.f,0.f};

    for (int k0 = 0; k0 < K; k0 += 64) {
        __syncthreads();
        // Stage A chunk [32 x 64]: float4 loads, packed v4h LDS stores.
        #pragma unroll
        for (int e = tid * 4; e < 32 * 64; e += 128 * 4) {
            int m = e >> 6, k = e & 63;
            float4 va = *(const float4*)&A[(size_t)m * K + k0 + k];
            v4h hv;
            hv[0] = (_Float16)va.x; hv[1] = (_Float16)va.y;
            hv[2] = (_Float16)va.z; hv[3] = (_Float16)va.w;
            *(v4h*)&As[m][k] = hv;
        }
        // Stage W chunk -> Bs[n][k] (transposed in LDS)
        if (!transB) {
            #pragma unroll
            for (int e = tid * 4; e < 64 * 64; e += 128 * 4) {
                int k = e >> 6, n = e & 63;              // float4 along n (coalesced)
                long long gn = nbase + n;
                float4 w = (gn < N) ? *(const float4*)&W[(size_t)(k0 + k) * N + gn]
                                    : zero4;
                Bs[n + 0][k] = (_Float16)w.x;
                Bs[n + 1][k] = (_Float16)w.y;
                Bs[n + 2][k] = (_Float16)w.z;
                Bs[n + 3][k] = (_Float16)w.w;
            }
        } else {
            #pragma unroll
            for (int e = tid * 4; e < 64 * 64; e += 128 * 4) {
                int n = e >> 6, k = e & 63;              // float4 along k (coalesced)
                long long gn = nbase + n;
                float4 w = (gn < N) ? *(const float4*)&W[(size_t)gn * K + k0 + k]
                                    : zero4;
                v4h hv;
                hv[0] = (_Float16)w.x; hv[1] = (_Float16)w.y;
                hv[2] = (_Float16)w.z; hv[3] = (_Float16)w.w;
                *(v4h*)&Bs[n][k] = hv;
            }
        }
        __syncthreads();

        #pragma unroll
        for (int kc = 0; kc < 64; kc += 32) {
            v16h a0, a1, bb;
            // A 16x32 f16 layout: lane r=row; elems 0..7 -> K=hi*8+j, 8..15 -> K=16+hi*8+j
            #pragma unroll
            for (int j = 0; j < 8; ++j) {
                a0[j]     = As[r][kc + hi * 8 + j];
                a0[8 + j] = As[r][kc + 16 + hi * 8 + j];
                a1[j]     = As[16 + r][kc + hi * 8 + j];
                a1[8 + j] = As[16 + r][kc + 16 + hi * 8 + j];
            }
            // B 32x16 f16 layout: lane r=col; elems 0..7 -> K=hi*8+j, 8..15 -> K=16+hi*8+j
            #pragma unroll
            for (int j = 0; j < 8; ++j) {
                bb[j]     = Bs[nl][kc + hi * 8 + j];
                bb[8 + j] = Bs[nl][kc + 16 + hi * 8 + j];
            }
            c0 = __builtin_amdgcn_wmma_f32_16x16x32_f16(false, a0, false, bb,
                                                        (short)0, c0, false, false);
            c1 = __builtin_amdgcn_wmma_f32_16x16x32_f16(false, a1, false, bb,
                                                        (short)0, c1, false, false);
        }
    }

    // Epilogue. C/D layout: VGPR j -> M = j + 8*hi; lane r -> N.
    long long gn = nbase + nl;
    if (gn < N) {
        float bv = bias ? bias[gn] : 0.f;
        #pragma unroll
        for (int j = 0; j < 8; ++j) {
            int m = j + 8 * hi;
            float v0 = c0[j] + bv;
            float v1 = c1[j] + bv;
            if (act == 1) { v0 = gelu_exact(v0); v1 = gelu_exact(v1); }
            if (res) {
                v0 += res[(size_t)m * N + gn];
                v1 += res[(size_t)(m + 16) * N + gn];
            }
            out[(size_t)m * N + gn]        = v0;
            out[(size_t)(m + 16) * N + gn] = v1;
        }
    }
}

// ---------------- LayerNorm: y[b,:] = (x-mean)*rsqrt(var+eps)*g + b. Block per row.
__global__ void ln_kernel(const float* __restrict__ x, const float* __restrict__ g,
                          const float* __restrict__ bta, float* __restrict__ y) {
    __shared__ float s1[128], s2[128];
    const int b = blockIdx.x, tid = threadIdx.x;
    const float* xr = x + (size_t)b * NS;
    float v0 = xr[tid], v1 = xr[tid + 128], v2 = xr[tid + 256];
    s1[tid] = v0 + v1 + v2;
    s2[tid] = v0 * v0 + v1 * v1 + v2 * v2;
    __syncthreads();
    for (int s = 64; s > 0; s >>= 1) {
        if (tid < s) { s1[tid] += s1[tid + s]; s2[tid] += s2[tid + s]; }
        __syncthreads();
    }
    float mean = s1[0] * (1.0f / NS);
    float var  = s2[0] * (1.0f / NS) - mean * mean;
    float inv  = rsqrtf(var + 1e-5f);
    float* yr = y + (size_t)b * NS;
    yr[tid]       = (v0 - mean) * inv * g[tid]       + bta[tid];
    yr[tid + 128] = (v1 - mean) * inv * g[tid + 128] + bta[tid + 128];
    yr[tid + 256] = (v2 - mean) * inv * g[tid + 256] + bta[tid + 256];
}

// ---------------- Attention (q-len 1): block per (b,h), 256 threads.
__global__ void attn_kernel(const float* __restrict__ q, const float* __restrict__ Kb,
                            const float* __restrict__ Vb, float* __restrict__ out, int T) {
    __shared__ float sc[1504];
    __shared__ float qs[64];
    __shared__ float red[256];
    __shared__ float part[8 * 64];
    const int tid = threadIdx.x;
    const int b = blockIdx.x / NH;
    const int h = blockIdx.x % NH;
    const size_t base = (size_t)b * T * NS + (size_t)h * DH;

    if (tid < 64) qs[tid] = q[(size_t)b * NS + h * DH + tid];
    __syncthreads();

    // scores: each thread streams one contiguous 256B K row per step (float4 loads)
    for (int t = tid; t < T; t += 256) {
        const float4* kr = (const float4*)(Kb + base + (size_t)t * NS);
        __builtin_prefetch(kr + 256 * (NS / 4), 0, 0);  // this thread's next row
        float d = 0.f;
        #pragma unroll
        for (int j = 0; j < 16; ++j) {
            float4 kk = kr[j];
            d += kk.x * qs[4 * j] + kk.y * qs[4 * j + 1] +
                 kk.z * qs[4 * j + 2] + kk.w * qs[4 * j + 3];
        }
        sc[t] = d * 0.125f;  // 1/sqrt(64)
    }
    __syncthreads();

    // max reduce
    float lm = -1e30f;
    for (int t = tid; t < T; t += 256) lm = fmaxf(lm, sc[t]);
    red[tid] = lm; __syncthreads();
    for (int s = 128; s > 0; s >>= 1) {
        if (tid < s) red[tid] = fmaxf(red[tid], red[tid + s]);
        __syncthreads();
    }
    float m = red[0];
    __syncthreads();

    // exp + sum
    float ls = 0.f;
    for (int t = tid; t < T; t += 256) {
        float e = expf(sc[t] - m);
        sc[t] = e;
        ls += e;
    }
    red[tid] = ls; __syncthreads();
    for (int s = 128; s > 0; s >>= 1) {
        if (tid < s) red[tid] += red[tid + s];
        __syncthreads();
    }
    float inv = 1.0f / red[0];
    __syncthreads();

    // P @ V : wave w owns t = w, w+8, ...; lane owns d = 2*lane, 2*lane+1
    const int wv = tid >> 5, lane = tid & 31;
    float ax = 0.f, ay = 0.f;
    for (int t = wv; t < T; t += 8) {
        float p = sc[t];
        const float2* vr = (const float2*)(Vb + base + (size_t)t * NS) + lane;
        float2 vvv = *vr;
        ax += p * vvv.x;
        ay += p * vvv.y;
    }
    part[wv * 64 + lane * 2]     = ax;
    part[wv * 64 + lane * 2 + 1] = ay;
    __syncthreads();
    if (tid < 64) {
        float s = 0.f;
        #pragma unroll
        for (int w = 0; w < 8; ++w) s += part[w * 64 + tid];
        out[(size_t)b * NS + h * DH + tid] = s * inv;
    }
}

// ---------------- KV cache append: dst[32,448,384] = concat(cache[32,447,384], new[32,384])
__global__ void append_kv_kernel(const float4* __restrict__ cache,
                                 const float4* __restrict__ nrow,
                                 float4* __restrict__ dst) {
    const size_t total = (size_t)BATCH * FULL_T * (NS / 4);
    for (size_t i = (size_t)blockIdx.x * blockDim.x + threadIdx.x; i < total;
         i += (size_t)gridDim.x * blockDim.x) {
        size_t c = i % (NS / 4);
        size_t r = (i / (NS / 4)) % FULL_T;
        size_t b = i / ((size_t)(NS / 4) * FULL_T);
        dst[i] = (r < CACHE_T) ? cache[(b * CACHE_T + r) * (NS / 4) + c]
                               : nrow[b * (NS / 4) + c];
    }
}

__global__ void copy_kernel(const float* __restrict__ src, float* __restrict__ dst, int n) {
    int i = blockIdx.x * blockDim.x + threadIdx.x;
    if (i < n) dst[i] = src[i];
}

// ---------------- Host orchestration ----------------
static inline void gemm(hipStream_t s, const float* A, const float* W, const float* bias,
                        const float* res, float* out, int K, int N, int transB, int act) {
    dim3 g((unsigned)((N + 63) / 64));
    wmma_gemm32_kernel<<<g, 128, 0, s>>>(A, W, bias, res, out, K, N, transB, act);
}

extern "C" void kernel_launch(void* const* d_in, const int* in_sizes, int n_in,
                              void* d_out, int out_size, void* d_ws, size_t ws_size,
                              hipStream_t stream) {
    const float* tok = (const float*)d_in[0];
    const float* ln_g = (const float*)d_in[17];
    const float* ln_b = (const float*)d_in[18];
    const float* Wq = (const float*)d_in[19];
    const float* bq = (const float*)d_in[20];
    const float* Wk = (const float*)d_in[21];
    const float* Wv = (const float*)d_in[22];
    const float* bv = (const float*)d_in[23];
    const float* Wo = (const float*)d_in[24];
    const float* bo = (const float*)d_in[25];
    const float* cln_g = (const float*)d_in[26];
    const float* cln_b = (const float*)d_in[27];
    const float* cWq = (const float*)d_in[28];
    const float* cbq = (const float*)d_in[29];
    const float* cWo = (const float*)d_in[30];
    const float* cbo = (const float*)d_in[31];
    const float* mln_g = (const float*)d_in[32];
    const float* mln_b = (const float*)d_in[33];
    const float* W1 = (const float*)d_in[34];
    const float* b1 = (const float*)d_in[35];
    const float* W2 = (const float*)d_in[36];
    const float* b2 = (const float*)d_in[37];
    const float* lnf_g = (const float*)d_in[38];
    const float* lnf_b = (const float*)d_in[39];
    const float* Wout = (const float*)d_in[40];

    // workspace layout (floats)
    float* ws  = (float*)d_ws;
    float* x   = ws;                 // [32,384]
    float* xln = x   + BATCH * NS;   // [32,384]
    float* qv  = xln + BATCH * NS;   // [32,384]
    float* kv  = qv  + BATCH * NS;   // [32,384]
    float* vv  = kv  + BATCH * NS;   // [32,384]
    float* att = vv  + BATCH * NS;   // [32,384]
    float* h1  = att + BATCH * NS;   // [32,1536]

    // output layout: logits | new_ks[4] | new_vs[4]
    float* logits = (float*)d_out;
    const size_t KVSZ = (size_t)BATCH * FULL_T * NS;
    float* newK = logits + (size_t)BATCH * VOCAB;
    float* newV = newK + NLAYER * KVSZ;

    const size_t W_s  = (size_t)NS * NS;   // per-layer weight stride
    const size_t W1_s = (size_t)NS * 4 * NS;

    copy_kernel<<<dim3((BATCH * NS + 255) / 256), 256, 0, stream>>>(tok, x, BATCH * NS);

    for (int i = 0; i < NLAYER; ++i) {
        const float* cross_k = (const float*)d_in[1 + 2 * i];
        const float* cross_v = (const float*)d_in[2 + 2 * i];
        const float* self_k  = (const float*)d_in[9 + 2 * i];
        const float* self_v  = (const float*)d_in[10 + 2 * i];
        float* outK = newK + (size_t)i * KVSZ;
        float* outV = newV + (size_t)i * KVSZ;

        // ----- self-attention -----
        ln_kernel<<<dim3(BATCH), 128, 0, stream>>>(x, ln_g + i * NS, ln_b + i * NS, xln);
        gemm(stream, xln, Wq + i * W_s, bq + i * NS, nullptr, qv, NS, NS, 0, 0);
        gemm(stream, xln, Wk + i * W_s, nullptr,     nullptr, kv, NS, NS, 0, 0);
        gemm(stream, xln, Wv + i * W_s, bv + i * NS, nullptr, vv, NS, NS, 0, 0);
        append_kv_kernel<<<dim3(1024), 256, 0, stream>>>(
            (const float4*)self_k, (const float4*)kv, (float4*)outK);
        append_kv_kernel<<<dim3(1024), 256, 0, stream>>>(
            (const float4*)self_v, (const float4*)vv, (float4*)outV);
        attn_kernel<<<dim3(BATCH * NH), 256, 0, stream>>>(qv, outK, outV, att, FULL_T);
        gemm(stream, att, Wo + i * W_s, bo + i * NS, x, x, NS, NS, 0, 0);

        // ----- cross-attention -----
        ln_kernel<<<dim3(BATCH), 128, 0, stream>>>(x, cln_g + i * NS, cln_b + i * NS, xln);
        gemm(stream, xln, cWq + i * W_s, cbq + i * NS, nullptr, qv, NS, NS, 0, 0);
        attn_kernel<<<dim3(BATCH * NH), 256, 0, stream>>>(qv, cross_k, cross_v, att, CROSS_T);
        gemm(stream, att, cWo + i * W_s, cbo + i * NS, x, x, NS, NS, 0, 0);

        // ----- MLP (exact GELU) -----
        ln_kernel<<<dim3(BATCH), 128, 0, stream>>>(x, mln_g + i * NS, mln_b + i * NS, xln);
        gemm(stream, xln, W1 + i * W1_s, b1 + i * 4 * NS, nullptr, h1, NS, 4 * NS, 0, 1);
        gemm(stream, h1, W2 + i * W1_s, b2 + i * NS, x, x, 4 * NS, NS, 0, 0);
    }

    // final LN + logits (Wout is [VOCAB, NS] -> transB GEMM)
    ln_kernel<<<dim3(BATCH), 128, 0, stream>>>(x, lnf_g, lnf_b, xln);
    gemm(stream, xln, Wout, nullptr, nullptr, logits, NS, VOCAB, 1, 0);
}
